// SlotKnowledgeAttention_82179904242089
// MI455X (gfx1250) — matrix operand order repaired and checked by reading, not compiled, not run on hardware
//
#include <hip/hip_runtime.h>

// ---------------------------------------------------------------------------
// SlotKnowledgeAttention for MI455X (gfx1250, wave32, WMMA bf16 f32-acc)
//
// Kernel 1 (k_proj_kernel): single pass over context (339MB, HBM-bound):
//   k[b,l,v,q] = [context|detect] @ W_k^T   (WMMA 16x16x32 bf16)
//   ctxsum[b,v,d] = sum_l context[b,l,v,d]  (fused via ds_add_f32 atomics)
//   W_k panel staged into LDS by the Tensor Data Mover (tensor_load_to_lds),
//   context rows software-prefetched (global_prefetch_b8) one tile ahead.
// Kernel 2 (slot_iter_kernel): 3 slot-attention iterations per batch, all
//   state in LDS; updates/gx/MLP GEMMs on WMMA.
// ---------------------------------------------------------------------------

typedef __attribute__((ext_vector_type(16))) __bf16 v16bf;
typedef __attribute__((ext_vector_type(8)))  float  v8f;
typedef unsigned int u32x4 __attribute__((ext_vector_type(4)));
typedef int          i32x4 __attribute__((ext_vector_type(4)));
typedef int          i32x8 __attribute__((ext_vector_type(8)));

union BF16x16 {
  v16bf v;
  unsigned short s[16];
  uint4 q[2];
};

__device__ __forceinline__ unsigned short f2bf(float f) {
  unsigned int x = __float_as_uint(f);
  return (unsigned short)((x + 0x7FFFu + ((x >> 16) & 1u)) >> 16);  // RNE
}

__device__ __forceinline__ v8f wmma_bf16(BF16x16 a, BF16x16 b, v8f c) {
  // (neg_a, A, neg_b, B, c_mod, C, reuse_a, reuse_b)
  return __builtin_amdgcn_wmma_f32_16x16x32_bf16(false, a.v, false, b.v,
                                                 (short)0, c, false, false);
}

// A-fragment (16x32 bf16, M x K) from an LDS bf16 tile [rows][strideE].
// ISA layout: lane<16 holds row m=lane, K = {0..7, 16..23}; lane>=16 holds
// K = {8..15, 24..31}. Each half is two contiguous 8-element (16B) runs.
__device__ __forceinline__ BF16x16 ldsA(const unsigned short* base, int strideE,
                                        int mt, int kc, int lane) {
  int m = lane & 15, half = lane >> 4;
  const unsigned short* p = base + (mt * 16 + m) * strideE + kc * 32 + half * 8;
  BF16x16 a;
  a.q[0] = *(const uint4*)(p);
  a.q[1] = *(const uint4*)(p + 16);
  return a;
}

// B-fragment (32x16 bf16, K x N) from an LDS bf16 tile stored [n][k]:
// lane holds column n=lane%16, K = (lane/16)*8 + {0..7, 16..23} (two 16B runs).
__device__ __forceinline__ BF16x16 ldsB(const unsigned short* base, int strideE,
                                        int nt, int kc, int lane) {
  const unsigned short* p =
      base + (nt * 16 + (lane & 15)) * strideE + kc * 32 + (lane >> 4) * 8;
  BF16x16 b;
  b.q[0] = *(const uint4*)(p);
  b.q[1] = *(const uint4*)(p + 16);
  return b;
}

// B-fragment from a global row-major weight W[N][K] (element (k,n)=W[n*ld+k]).
// Per lane the 16 K-values are two 8-element runs, k0+{0..7, 16..23}; 16 lanes hit 16 rows.
__device__ __forceinline__ BF16x16 gB(const float* __restrict__ W, int ldW,
                                      int nt, int kc, int nMax, int kMax,
                                      int lane) {
  int n = nt * 16 + (lane & 15);
  int k0 = kc * 32 + (lane >> 4) * 8;
  BF16x16 b;
#pragma unroll
  for (int t = 0; t < 16; ++t) {
    int k = k0 + ((t < 8) ? t : t + 8);
    float v = (n < nMax && k < kMax) ? W[(size_t)n * ldW + k] : 0.f;
    b.s[t] = f2bf(v);
  }
  return b;
}

// B-fragment from a global [K][N] matrix (element (k,n)=M[k*ld+n]); loads are
// lane-coalesced (16 consecutive n per instruction).
__device__ __forceinline__ BF16x16 gBT(const float* __restrict__ M, int ldM,
                                       int nt, int kc, int kMax, int lane) {
  int n = nt * 16 + (lane & 15);
  int k0 = kc * 32 + (lane >> 4) * 8;
  BF16x16 b;
#pragma unroll
  for (int t = 0; t < 16; ++t) {
    int k = k0 + ((t < 8) ? t : t + 8);
    float v = (k < kMax) ? M[(size_t)k * ldM + n] : 0.f;
    b.s[t] = f2bf(v);
  }
  return b;
}

// ---------------------------------------------------------------------------
// Kernel 1: k projection + fused ctxsum.
// Grid: 128 batches x 3 parts. Rows are processed v-major (r = v*36 + l) so a
// part of 432 rows = 12 whole v's -> ctxsum accumulation is block-local.
// ---------------------------------------------------------------------------
#define WK_STRIDE 840   // 832 bf16 + pad (16*840*2 % 16B aligned, bank-spread)
#define CS_STRIDE 516

__global__ __launch_bounds__(256) void k_proj_kernel(
    const float* __restrict__ context, const float* __restrict__ detect,
    const float* __restrict__ Wk, float* __restrict__ k_out,
    float* __restrict__ ctxsum_out) {
  extern __shared__ unsigned char smem[];
  unsigned short* wk_lds = (unsigned short*)smem;              // [48][840] bf16
  float* csum = (float*)(smem + 48 * WK_STRIDE * 2);           // [12][516] f32
  float* wk_f32 = (float*)(smem + 48 * WK_STRIDE * 2 + 12 * CS_STRIDE * 4);
                                                               // [36*812] f32

  int tid = threadIdx.x;
  int b = blockIdx.x / 3, part = blockIdx.x % 3;
  int lane = tid & 31, wid = tid >> 5;
  int m = lane & 15, half = lane >> 4;

  // --- TDM: DMA the whole W_k panel (36x812 f32, row-major) into LDS. ---
  // D# per CDNA5 ISA: group0 = {count=1 | lds_addr |
  // global_addr | type=2}; group1 = {mask=0, data_size=4B, tensor_dim0=812,
  // tensor_dim1=36, tile_dim0=812, tile_dim1=36, dim0_stride=812}.
  if (wid == 0) {
    unsigned long long ga = (unsigned long long)(const void*)Wk;
    unsigned int lds_off = (unsigned int)(size_t)(void*)wk_f32;  // low 32b
    u32x4 g0;
    g0[0] = 1u;                                   // count=1 (valid descriptor)
    g0[1] = lds_off;                              // lds_addr [63:32]
    g0[2] = (unsigned int)ga;                     // global_addr [95:64]
    g0[3] = ((unsigned int)(ga >> 32) & 0x01FFFFFFu) | 0x80000000u;  // type=2
    i32x8 g1;
    g1[0] = (int)(2u << 16);                      // data_size = 4 bytes
    g1[1] = (int)(812u << 16);                    // tensor_dim0[15:0] @ 48
    g1[2] = (int)(36u << 16);                     // dim0 hi=0 | tensor_dim1
    g1[3] = (int)(812u << 16);                    // dim1 hi=0 | tile_dim0
    g1[4] = 36;                                   // tile_dim1 (tile_dim2=0)
    g1[5] = 812;                                  // tensor_dim0_stride lo32
    g1[6] = 0;
    g1[7] = 0;
    i32x4 z4 = {0, 0, 0, 0};
#if __clang_major__ >= 23
    i32x8 z8 = {0, 0, 0, 0, 0, 0, 0, 0};
    __builtin_amdgcn_tensor_load_to_lds(g0, g1, z4, z4, z8, 0);
#else
    __builtin_amdgcn_tensor_load_to_lds(g0, g1, z4, z4, 0);
#endif
    __builtin_amdgcn_s_wait_tensorcnt(0);
  }
  for (int idx = tid; idx < 12 * CS_STRIDE; idx += 256) csum[idx] = 0.f;
  __syncthreads();

  // Convert staged W_k to bf16 [n=q][k=c], zero-padded to 48x832.
  for (int idx = tid; idx < 48 * 832; idx += 256) {
    int n = idx / 832, k = idx % 832;
    float v = (n < 36 && k < 812) ? wk_f32[n * 812 + k] : 0.f;
    wk_lds[n * WK_STRIDE + k] = f2bf(v);
  }
  __syncthreads();

  int v0 = part * 12;
  for (int tile = wid; tile < 27; tile += 8) {
    int R = part * 432 + tile * 16 + m;  // v-major row
    int v = R / 36, l = R % 36;
    const float* ctxrow = context + (((size_t)b * 36 + l) * 36 + v) * 512;
    const float* detrow = detect + ((size_t)b * 36 + l) * 300;

    // Prefetch next tile's context row (one tile ahead of the WMMA consume).
    if (tile + 8 < 27) {
      int R2 = part * 432 + (tile + 8) * 16 + m;
      const float* nrow =
          context + (((size_t)b * 36 + (R2 % 36)) * 36 + (R2 / 36)) * 512;
      __builtin_prefetch(nrow, 0, 3);                   // global_prefetch_b8
      __builtin_prefetch(nrow + 256 + half * 16, 0, 3);
    }

    v8f acc0 = {}, acc1 = {}, acc2 = {};
    for (int kc = 0; kc < 26; ++kc) {
      BF16x16 a;
      if (kc < 16) {  // context columns: vector loads + fused ds_add reduction
        const float* p = ctxrow + kc * 32 + half * 8;
        float fv[16];
        *(float4*)(fv + 0)  = *(const float4*)(p + 0);
        *(float4*)(fv + 4)  = *(const float4*)(p + 4);
        *(float4*)(fv + 8)  = *(const float4*)(p + 16);
        *(float4*)(fv + 12) = *(const float4*)(p + 20);
        float* cr = csum + (v - v0) * CS_STRIDE + kc * 32 + half * 8;
#pragma unroll
        for (int t = 0; t < 16; ++t) {
          a.s[t] = f2bf(fv[t]);
          atomicAdd(cr + ((t < 8) ? t : t + 8), fv[t]);  // ds_add_f32
        }
      } else {  // detect columns (broadcast over v), tail-guarded
        int c0 = kc * 32 - 512 + half * 8;
#pragma unroll
        for (int t = 0; t < 8; ++t) {
          int ca = c0 + t, cb = c0 + 16 + t;
          a.s[t]     = f2bf(ca < 300 ? detrow[ca] : 0.f);
          a.s[8 + t] = f2bf(cb < 300 ? detrow[cb] : 0.f);
        }
      }
      acc0 = wmma_bf16(a, ldsB(wk_lds, WK_STRIDE, 0, kc, lane), acc0);
      acc1 = wmma_bf16(a, ldsB(wk_lds, WK_STRIDE, 1, kc, lane), acc1);
      acc2 = wmma_bf16(a, ldsB(wk_lds, WK_STRIDE, 2, kc, lane), acc2);
    }

    // Store k[b,l,v,q]; C/D layout: vgpr j -> M=j+8*half, N=lane%16.
#pragma unroll
    for (int j = 0; j < 8; ++j) {
      int Rr = part * 432 + tile * 16 + j + 8 * half;
      int vv = Rr / 36, ll = Rr % 36;
      float* krow = k_out + (((size_t)b * 36 + ll) * 36 + vv) * 36;
      int n = lane & 15;
      krow[n] = acc0[j];
      krow[16 + n] = acc1[j];
      if (n < 4) krow[32 + n] = acc2[j];
    }
  }
  __syncthreads();
  for (int idx = tid; idx < 12 * 512; idx += 256) {
    int vv = idx >> 9, d = idx & 511;
    ctxsum_out[((size_t)b * 36 + v0 + vv) * 512 + d] = csum[vv * CS_STRIDE + d];
  }
}

// ---------------------------------------------------------------------------
// Kernel 2: 3 slot-attention iterations, one block per batch (8 waves).
// ---------------------------------------------------------------------------
#define SS 40    // f32 row stride for 36-wide matrices
#define UPS 528  // bf16 row stride for 512-wide tiles (48 rows)
#define ATS 64   // bf16 row stride for 36-wide tiles  (48 rows)

__global__ __launch_bounds__(256) void slot_iter_kernel(
    const float* __restrict__ pano, const float* __restrict__ Wq,
    const float* __restrict__ k_in, const float* __restrict__ ctxsum,
    const float* __restrict__ knowledge, const float* __restrict__ Wih,
    const float* __restrict__ Whh, const float* __restrict__ bih,
    const float* __restrict__ bhh, const float* __restrict__ lsg,
    const float* __restrict__ lsb, const float* __restrict__ lfg,
    const float* __restrict__ lfb, const float* __restrict__ W1,
    const float* __restrict__ b1, const float* __restrict__ W2,
    const float* __restrict__ b2, float* __restrict__ out_slots,
    float* __restrict__ out_attn) {
  extern __shared__ unsigned char smem[];
  float* f_slots = (float*)smem;            // [36][40]
  float* f_dots  = f_slots + 36 * SS;       // [36][40] dots -> attn (f32)
  float* f_gnew  = f_dots + 36 * SS;        // [36][40]
  float* f_gx    = f_gnew + 36 * SS;        // [36][112] gates from x
  float* f_gxk   = f_gx + 36 * 112;         // [36][112] knowledge part + bih
  float* f_mu    = f_gxk + 36 * 112;        // [40]
  float* f_rs    = f_mu + 40;               // [40]
  float* f_ln    = f_rs + 40;               // [40] LN column sums
  float* f_qw    = f_ln + 40;               // [40] summed query
  unsigned short* u_upd  = (unsigned short*)(f_qw + 40);  // [48][528] bf16
  unsigned short* u_attn = u_upd + 48 * UPS;              // [48][64]  bf16
  unsigned short* u_ff   = u_attn + 48 * ATS;             // [48][64]  bf16

  int b = blockIdx.x, tid = threadIdx.x;
  int lane = tid & 31, wid = tid >> 5, half = lane >> 4;

  // ---- one-time init ----
  for (int i = tid; i < 48 * UPS; i += 256) u_upd[i] = 0;
  for (int i = tid; i < 48 * ATS; i += 256) { u_attn[i] = 0; u_ff[i] = 0; }
  for (int p = tid; p < 1296; p += 256)
    f_slots[(p / 36) * SS + (p % 36)] = pano[p];
  // gxk = knowledge @ Wih[:,512:624]^T + bih  (one-time, VALU)
  for (int t = tid; t < 36 * 108; t += 256) {
    int i = t / 108, g = t % 108;
    const float* kn = knowledge + ((size_t)b * 36 + i) * 112;
    const float* w = Wih + (size_t)g * 624 + 512;
    float s = 0.f;
    for (int c = 0; c < 112; ++c) s += kn[c] * w[c];
    f_gxk[i * 112 + g] = s + bih[g];
  }
  __syncthreads();

  for (int it = 0; it < 3; ++it) {
    // A) per-row LN stats of slots
    if (tid < 36) {
      const float* r = f_slots + tid * SS;
      float mu = 0.f;
      for (int e = 0; e < 36; ++e) mu += r[e];
      mu *= (1.f / 36.f);
      float var = 0.f;
      for (int e = 0; e < 36; ++e) { float d = r[e] - mu; var += d * d; }
      var *= (1.f / 36.f);
      f_mu[tid] = mu;
      f_rs[tid] = rsqrtf(var + 1e-5f);
    }
    __syncthreads();
    // B) column sums of LN(slots): sum_i q collapses the query GEMM
    if (tid < 36) {
      float s = 0.f;
      for (int i = 0; i < 36; ++i)
        s += (f_slots[i * SS + tid] - f_mu[i]) * f_rs[i];
      f_ln[tid] = lsg[tid] * s + 36.f * lsb[tid];
    }
    __syncthreads();
    // C) qw = lnsum @ W_q^T   (36x36 matvec)
    if (tid < 36) {
      const float* w = Wq + tid * 36;
      float s = 0.f;
      for (int e = 0; e < 36; ++e) s += f_ln[e] * w[e];
      f_qw[tid] = s;
    }
    __syncthreads();
    // D) dots[l,v] = (1/6) * qw . k[b,l,v,:]
    for (int p = tid; p < 1296; p += 256) {
      int l = p / 36, v = p % 36;
      const float* kr = k_in + (((size_t)b * 36 + l) * 36 + v) * 36;
      float s = 0.f;
      for (int q = 0; q < 36; ++q) s += f_qw[q] * kr[q];
      f_dots[l * SS + v] = s * (1.f / 6.f);
    }
    __syncthreads();
    // E) softmax over views + emit attn (global f32 + LDS bf16 A-tile)
    if (tid < 36) {
      float* r = f_dots + tid * SS;
      float mx = r[0];
      for (int v = 1; v < 36; ++v) mx = fmaxf(mx, r[v]);
      float s = 0.f;
      for (int v = 0; v < 36; ++v) { float e = __expf(r[v] - mx); r[v] = e; s += e; }
      float inv = 1.f / s;
      float* oa = out_attn + (((size_t)it * 128 + b) * 36 + tid) * 36;
      for (int v = 0; v < 36; ++v) {
        float a = r[v] * inv;
        r[v] = a;
        oa[v] = a;
        u_attn[tid * ATS + v] = f2bf(a);
      }
    }
    __syncthreads();
    // F) updates = attn @ ctxsum   (M=48, N=512, K=64) -> u_upd bf16
    for (int job = wid; job < 96; job += 8) {
      int mt = job >> 5, nt = job & 31;
      v8f acc = {};
      for (int kc = 0; kc < 2; ++kc)
        acc = wmma_bf16(ldsA(u_attn, ATS, mt, kc, lane),
                        gBT(ctxsum + (size_t)b * 36 * 512, 512, nt, kc, 36, lane),
                        acc);
      int n = nt * 16 + (lane & 15);
#pragma unroll
      for (int j = 0; j < 8; ++j) {
        int row = mt * 16 + j + 8 * half;
        if (row < 36) u_upd[row * UPS + n] = f2bf(acc[j]);
      }
    }
    __syncthreads();
    // G) gx = updates @ Wih[:,:512]^T + gxk   (M=48, N=112, K=512)
    for (int job = wid; job < 21; job += 8) {
      int mt = job / 7, nt = job % 7;
      v8f acc = {};
      for (int kc = 0; kc < 16; ++kc)
        acc = wmma_bf16(ldsA(u_upd, UPS, mt, kc, lane),
                        gB(Wih, 624, nt, kc, 108, 512, lane), acc);
      int n = nt * 16 + (lane & 15);
      if (n < 108) {
#pragma unroll
        for (int j = 0; j < 8; ++j) {
          int row = mt * 16 + j + 8 * half;
          if (row < 36) f_gx[row * 112 + n] = acc[j] + f_gxk[row * 112 + n];
        }
      }
    }
    __syncthreads();
    // H) GRU cell (gates r,z,n); gh computed on VALU (K=36 tiny)
    for (int p = tid; p < 1296; p += 256) {
      int i = p / 36, u = p % 36;
      const float* hrow = f_slots + i * SS;
      const float* wr = Whh + u * 36;
      const float* wz = Whh + (36 + u) * 36;
      const float* wn = Whh + (72 + u) * 36;
      float gr = bhh[u], gz = bhh[36 + u], gn = bhh[72 + u];
      for (int e = 0; e < 36; ++e) {
        float h = hrow[e];
        gr += h * wr[e]; gz += h * wz[e]; gn += h * wn[e];
      }
      float xr = f_gx[i * 112 + u], xz = f_gx[i * 112 + 36 + u],
            xn = f_gx[i * 112 + 72 + u];
      float r = 1.f / (1.f + __expf(-(xr + gr)));
      float z = 1.f / (1.f + __expf(-(xz + gz)));
      float nn = tanhf(xn + r * gn);
      f_gnew[i * SS + u] = (1.f - z) * nn + z * hrow[u];
    }
    __syncthreads();
    // I) ff = LN(gnew) -> bf16 A-tile
    if (tid < 36) {
      const float* r = f_gnew + tid * SS;
      float mu = 0.f;
      for (int e = 0; e < 36; ++e) mu += r[e];
      mu *= (1.f / 36.f);
      float var = 0.f;
      for (int e = 0; e < 36; ++e) { float d = r[e] - mu; var += d * d; }
      var *= (1.f / 36.f);
      float rs = rsqrtf(var + 1e-5f);
      for (int e = 0; e < 36; ++e)
        u_ff[tid * ATS + e] = f2bf((r[e] - mu) * rs * lfg[e] + lfb[e]);
    }
    __syncthreads();
    // J) hidden = relu(ff @ W1^T + b1)   (M=48, N=512, K=64) -> reuse u_upd
    for (int job = wid; job < 96; job += 8) {
      int mt = job >> 5, nt = job & 31;
      v8f acc = {};
      for (int kc = 0; kc < 2; ++kc)
        acc = wmma_bf16(ldsA(u_ff, ATS, mt, kc, lane),
                        gB(W1, 36, nt, kc, 512, 36, lane), acc);
      int n = nt * 16 + (lane & 15);
      float bias = b1[n];
#pragma unroll
      for (int j = 0; j < 8; ++j) {
        int row = mt * 16 + j + 8 * half;
        if (row < 36) u_upd[row * UPS + n] = f2bf(fmaxf(acc[j] + bias, 0.f));
      }
    }
    __syncthreads();
    // K) slots = gnew + hidden @ W2^T + b2   (M=48, N=48, K=512)
    for (int job = wid; job < 9; job += 8) {
      int mt = job / 3, nt = job % 3;
      v8f acc = {};
      for (int kc = 0; kc < 16; ++kc)
        acc = wmma_bf16(ldsA(u_upd, UPS, mt, kc, lane),
                        gB(W2, 512, nt, kc, 36, 512, lane), acc);
      int n = nt * 16 + (lane & 15);
      if (n < 36) {
        float bias = b2[n];
#pragma unroll
        for (int j = 0; j < 8; ++j) {
          int row = mt * 16 + j + 8 * half;
          if (row < 36)
            f_slots[row * SS + n] = f_gnew[row * SS + n] + acc[j] + bias;
        }
      }
    }
    __syncthreads();
  }
  // final slots -> out
  for (int p = tid; p < 1296; p += 256)
    out_slots[(size_t)b * 1296 + p] = f_slots[(p / 36) * SS + (p % 36)];
}

// ---------------------------------------------------------------------------
extern "C" void kernel_launch(void* const* d_in, const int* in_sizes, int n_in,
                              void* d_out, int out_size, void* d_ws,
                              size_t ws_size, hipStream_t stream) {
  (void)in_sizes; (void)n_in; (void)out_size; (void)ws_size;
  const float* context   = (const float*)d_in[1];
  const float* detect    = (const float*)d_in[3];
  const float* knowledge = (const float*)d_in[4];
  const float* pano      = (const float*)d_in[5];
  const float* Wq        = (const float*)d_in[6];
  const float* Wk        = (const float*)d_in[7];
  const float* Wih       = (const float*)d_in[8];
  const float* Whh       = (const float*)d_in[9];
  const float* bih       = (const float*)d_in[10];
  const float* bhh       = (const float*)d_in[11];
  const float* lsg       = (const float*)d_in[12];
  const float* lsb       = (const float*)d_in[13];
  const float* lfg       = (const float*)d_in[14];
  const float* lfb       = (const float*)d_in[15];
  const float* W1        = (const float*)d_in[16];
  const float* b1        = (const float*)d_in[17];
  const float* W2        = (const float*)d_in[18];
  const float* b2        = (const float*)d_in[19];

  float* k_ws      = (float*)d_ws;                 // 128*36*36*36 f32
  float* ctxsum_ws = k_ws + (size_t)128 * 36 * 36 * 36;  // 128*36*512 f32

  float* out_slots = (float*)d_out;                // (128,36,36)
  float* out_attn  = out_slots + (size_t)128 * 36 * 36;  // (3,128,36,36)

  const int smem1 = 48 * WK_STRIDE * 2 + 12 * CS_STRIDE * 4 +
                    36 * 812 * 4;                             // 222,336 B
  const int smem2 =
      (3 * 36 * SS + 2 * 36 * 112 + 4 * 40) * 4 +
      (48 * UPS + 2 * 48 * ATS) * 2;                          // 113,152 B

  k_proj_kernel<<<dim3(384), dim3(256), smem1, stream>>>(
      context, detect, Wk, k_ws, ctxsum_ws);
  slot_iter_kernel<<<dim3(128), dim3(256), smem2, stream>>>(
      pano, Wq, k_ws, ctxsum_ws, knowledge, Wih, Whh, bih, bhh, lsg, lsb, lfg,
      lfb, W1, b1, W2, b2, out_slots, out_attn);
}
